// formerBlock_62723702391248
// MI455X (gfx1250) — compile-verified, matrix operand order checked by reading, not run on hardware
//
#include <hip/hip_runtime.h>
#include <hip/hip_bf16.h>
#include <cstdint>
#include <cstddef>

// ---------------------------------------------------------------------------
// MI455X (gfx1250) implementation of the wavelet-window-attention block.
// All matmuls (qkv / attention / proj / aux_pw / attn_out / ffn_in / ffn_out)
// run on v_wmma_f32_16x16x32_f16. Activations between GEMMs are f16 in
// fragment-friendly row-major layouts. The per-patch rfft2*W*irfft2 is
// replaced by an exact 8x8 circular convolution with irfft2(W).
// GEMM waves compute 16x64 strips; K-chunks are ping-pong double-buffered
// (unroll-by-2, KC always even here) so global loads overlap the WMMA pipe
// with no register-rotation moves.
// ---------------------------------------------------------------------------

typedef __attribute__((ext_vector_type(16))) _Float16 v16h;
typedef __attribute__((ext_vector_type(8)))  _Float16 v8h;
typedef __attribute__((ext_vector_type(8)))  float    v8f;

#define DEVINL __device__ __forceinline__

DEVINL float gelu_f(float v) { return 0.5f * v * (1.0f + erff(v * 0.70710678118654752f)); }

// Load a 16x32 f16 WMMA A/B fragment from 16 row-major rows (stride in halves).
// Lane layout (ISA 7.12.2): lanes 0-15 row=lane, K in {0..7,16..23};
// lanes 16-31 row=lane-16, K in {8..15,24..31}. Two contiguous 16B loads/lane.
DEVINL v16h load_frag(const _Float16* p, int stride) {
  const int lane = threadIdx.x & 31;
  const _Float16* r = p + (size_t)(lane & 15) * stride;
  const int hi = (lane & 16) ? 8 : 0;
  v8h a0 = *(const v8h*)(r + hi);
  v8h a1 = *(const v8h*)(r + 16 + hi);
  v16h f;
#pragma unroll
  for (int e = 0; e < 8; ++e) { f[e] = a0[e]; f[8 + e] = a1[e]; }
  return f;
}

// Load a pre-packed B fragment (512 halves per 32x16 tile, lane-major).
DEVINL v16h load_packB(const _Float16* bp) {
  const int lane = threadIdx.x & 31;
  const _Float16* r = bp + lane * 16;
  v8h b0 = *(const v8h*)(r);
  v8h b1 = *(const v8h*)(r + 8);
  v16h f;
#pragma unroll
  for (int e = 0; e < 8; ++e) { f[e] = b0[e]; f[8 + e] = b1[e]; }
  return f;
}

// ---------------------------------------------------------------------------
// Weight packing: W is [N_out][K_in] fp32 (PyTorch layout, used as x @ W^T).
// Pack B[k][n] = W[n][k] into per-tile fragment layout.
// ---------------------------------------------------------------------------
__global__ void k_pack_b(const float* __restrict__ W, _Float16* __restrict__ Bp,
                         int K, int N) {
  const int idx = blockIdx.x * blockDim.x + threadIdx.x;
  const int NT = N >> 4;
  const int total = (K >> 5) * NT * 512;
  if (idx >= total) return;
  const int e = idx & 15;
  const int lane = (idx >> 4) & 31;
  const int tile = idx >> 9;
  const int nt = tile % NT;
  const int kc = tile / NT;
  const int hi = (lane & 16) ? 8 : 0;
  const int k = kc * 32 + ((e < 8) ? (hi + e) : (8 + hi + e));
  const int n = nt * 16 + (lane & 15);
  Bp[idx] = (_Float16)W[(size_t)n * K + k];
}

// ---------------------------------------------------------------------------
// Generic WMMA GEMM: C[M,N] = A[M,K] (f16 row-major) * packedB (+bias).
// 8 waves / block; each wave owns a 16(M) x 64(N) strip. KC must be even:
// the K loop advances two chunks per iteration with two disjoint fragment
// buffer sets (ping-pong), so loads for one phase overlap WMMAs of the other
// and no cross-iteration register copies are needed.
// ---------------------------------------------------------------------------
__global__ __launch_bounds__(256) void k_gemm(
    const _Float16* __restrict__ A, const _Float16* __restrict__ Bp,
    const float* __restrict__ bias, float* __restrict__ Cf,
    _Float16* __restrict__ Ch, int M, int N, int K) {
  (void)M;
  const int NT = N >> 4;   // 16-wide tiles
  const int NS = NT >> 2;  // 64-wide strips
  const int KC = K >> 5;   // even for all GEMMs in this graph
  const int strip = blockIdx.x * 8 + (threadIdx.x >> 5);
  const int mt = strip / NS;
  const int ns = strip % NS;
  const int lane = threadIdx.x & 31;
  const _Float16* Abase = A + (size_t)mt * 16 * K;
  const _Float16* Bbase = Bp + (size_t)(ns * 4) * 512;
  const size_t bstride = (size_t)NT * 512;  // packed halves per K-chunk

  const v8f z = {};
  v8f acc[4] = {z, z, z, z};

  // phase-A buffers: chunk 0
  v16h afA = load_frag(Abase, K);
  v16h bfA[4];
#pragma unroll
  for (int q = 0; q < 4; ++q) bfA[q] = load_packB(Bbase + (size_t)q * 512);

  for (int kc = 0; kc < KC; kc += 2) {
    // phase-B buffers: chunk kc+1 (always exists, KC even)
    if (kc + 3 < KC) __builtin_prefetch(Abase + (kc + 3) * 32, 0, 1);
    v16h afB = load_frag(Abase + (kc + 1) * 32, K);
    v16h bfB[4];
#pragma unroll
    for (int q = 0; q < 4; ++q)
      bfB[q] = load_packB(Bbase + (size_t)(kc + 1) * bstride + (size_t)q * 512);

    // WMMAs on phase-A while phase-B loads are in flight
#pragma unroll
    for (int q = 0; q < 4; ++q)
      acc[q] = __builtin_amdgcn_wmma_f32_16x16x32_f16(false, afA, false, bfA[q],
                                                      (short)0, acc[q], false, false);

    // refill phase-A with chunk kc+2 while phase-B WMMAs run
    if (kc + 2 < KC) {
      afA = load_frag(Abase + (kc + 2) * 32, K);
#pragma unroll
      for (int q = 0; q < 4; ++q)
        bfA[q] = load_packB(Bbase + (size_t)(kc + 2) * bstride + (size_t)q * 512);
    }

#pragma unroll
    for (int q = 0; q < 4; ++q)
      acc[q] = __builtin_amdgcn_wmma_f32_16x16x32_f16(false, afB, false, bfB[q],
                                                      (short)0, acc[q], false, false);
  }

  const int mrow = mt * 16 + ((lane & 16) ? 8 : 0);
#pragma unroll
  for (int q = 0; q < 4; ++q) {
    const int n = (ns * 4 + q) * 16 + (lane & 15);
    const float bv = bias ? bias[n] : 0.0f;
#pragma unroll
    for (int r = 0; r < 8; ++r) {
      const size_t o = (size_t)(mrow + r) * N + n;
      const float v = acc[q][r] + bv;
      if (Ch) Ch[o] = (_Float16)v;
      else    Cf[o] = v;
    }
  }
}

// ---------------------------------------------------------------------------
// DWT: x (4,64,256,256) -> ll (4,64,128,128), high (4,192,128,128)
// ---------------------------------------------------------------------------
__global__ __launch_bounds__(256) void k_dwt(const float* __restrict__ x,
                                             float* __restrict__ ll,
                                             float* __restrict__ high) {
  const int idx = blockIdx.x * blockDim.x + threadIdx.x;
  if (idx >= 4 * 64 * 128 * 128) return;
  const int j = idx & 127, i = (idx >> 7) & 127, c = (idx >> 14) & 63, b = idx >> 20;
  const float* xb = x + (((size_t)b * 64 + c) * 256 + 2 * i) * 256 + 2 * j;
  const float x1 = xb[0] * 0.5f;
  const float x3 = xb[1] * 0.5f;
  const float x2 = xb[256] * 0.5f;
  const float x4 = xb[257] * 0.5f;
  ll[idx] = x1 + x2 + x3 + x4;
  const size_t hb = (((size_t)b * 192 + c) * 128 + i) * 128 + j;
  high[hb]                  = -x1 - x2 + x3 + x4;  // HL
  high[hb + 64 * 128 * 128] = -x1 + x2 - x3 + x4;  // LH
  high[hb + 128 * 128 * 128] = x1 - x2 - x3 + x4;  // HH
}

// ---------------------------------------------------------------------------
// Per-pixel LN over 192 channels + roll(-4,-4) + 8x8 window partition -> f16
// hn[win][tok][192].  One block per (win,tok).
// ---------------------------------------------------------------------------
__global__ __launch_bounds__(256) void k_ln_hn(const float* __restrict__ high,
                                               const float* __restrict__ w,
                                               const float* __restrict__ bse,
                                               _Float16* __restrict__ hn) {
  __shared__ float red[256];
  __shared__ float smu, srs;
  const int blk = blockIdx.x;
  const int tok = blk & 63, win = blk >> 6;
  const int b = win >> 8, wr = (win >> 4) & 15, wc = win & 15;
  const int i = wr * 8 + (tok >> 3), j = wc * 8 + (tok & 7);
  const int sh = (i + 4) & 127, sw = (j + 4) & 127;
  const int c = threadIdx.x;
  float v = 0.0f;
  if (c < 192) v = high[(((size_t)b * 192 + c) * 128 + sh) * 128 + sw];
  red[c] = v;
  __syncthreads();
  for (int s = 128; s > 0; s >>= 1) { if (c < s) red[c] += red[c + s]; __syncthreads(); }
  if (c == 0) smu = red[0] * (1.0f / 192.0f);
  __syncthreads();
  const float mu = smu;
  const float d = (c < 192) ? (v - mu) : 0.0f;
  red[c] = d * d;
  __syncthreads();
  for (int s = 128; s > 0; s >>= 1) { if (c < s) red[c] += red[c + s]; __syncthreads(); }
  if (c == 0) srs = rsqrtf(red[0] * (1.0f / 192.0f) + 1e-6f);
  __syncthreads();
  if (c < 192) {
    const float rs = srs;
    hn[((size_t)win * 64 + tok) * 192 + c] = (_Float16)(d * rs * w[c] + bse[c]);
  }
}

// Relative position bias table expansion: bias66[h][q][k]
__global__ void k_relbias(const float* __restrict__ table,
                          const int* __restrict__ ridx,
                          float* __restrict__ bias66) {
  const int t = blockIdx.x * blockDim.x + threadIdx.x;
  if (t >= 6 * 4096) return;
  const int h = t / 4096, qk = t % 4096;
  bias66[t] = table[ridx[qk] * 6 + h];
}

// ---------------------------------------------------------------------------
// Windowed attention, one wave per (window, head). S = QK^T via WMMA,
// softmax in-register (+bias), P staged through LDS to re-fragment, O = PV.
// ---------------------------------------------------------------------------
__global__ __launch_bounds__(32) void k_attn(const _Float16* __restrict__ qkv,
                                             const float* __restrict__ bias66,
                                             _Float16* __restrict__ outp) {
  __shared__ __align__(16) _Float16 Pl[64 * 64];
  __shared__ __align__(16) _Float16 Vt[32 * 64];
  const int win = blockIdx.x;
  const int head = blockIdx.y;
  const int lane = threadIdx.x;
  const _Float16* base = qkv + (size_t)win * 64 * 576;
  const _Float16* Qm = base + head * 32;
  const _Float16* Km = base + 192 + head * 32;
  const _Float16* Vm = base + 384 + head * 32;

  // Stage V^T into LDS: Vt[d][t]
#pragma unroll
  for (int tt = 0; tt < 2; ++tt) {
    const int t = lane * 2 + tt;
    const _Float16* vr = Vm + (size_t)t * 576;
#pragma unroll
    for (int cg = 0; cg < 4; ++cg) {
      v8h v = *(const v8h*)(vr + cg * 8);
#pragma unroll
      for (int e = 0; e < 8; ++e) Vt[(cg * 8 + e) * 64 + t] = v[e];
    }
  }

  // S = Q K^T (64x64, K=32 -> one WMMA chunk per tile pair)
  v16h qa[4], kb[4];
#pragma unroll
  for (int mt = 0; mt < 4; ++mt) qa[mt] = load_frag(Qm + (size_t)mt * 16 * 576, 576);
#pragma unroll
  for (int nt = 0; nt < 4; ++nt) kb[nt] = load_frag(Km + (size_t)nt * 16 * 576, 576);

  const v8f vzero = {};
  v8f s[4][4];
#pragma unroll
  for (int mt = 0; mt < 4; ++mt)
#pragma unroll
    for (int nt = 0; nt < 4; ++nt)
      s[mt][nt] = __builtin_amdgcn_wmma_f32_16x16x32_f16(false, qa[mt], false, kb[nt],
                                                         (short)0, vzero, false, false);

  // softmax rows (+rel-pos bias), write P (f16) to LDS
  const float scale = 0.17677669529663687f;  // 1/sqrt(32)
  const float* bb = bias66 + head * 4096;
  const int n0 = lane & 15;
  const int mb = (lane & 16) ? 8 : 0;
#pragma unroll
  for (int mt = 0; mt < 4; ++mt) {
#pragma unroll
    for (int r = 0; r < 8; ++r) {
      const int m = mt * 16 + mb + r;
      float v0 = s[mt][0][r] * scale + bb[m * 64 + 0 + n0];
      float v1 = s[mt][1][r] * scale + bb[m * 64 + 16 + n0];
      float v2 = s[mt][2][r] * scale + bb[m * 64 + 32 + n0];
      float v3 = s[mt][3][r] * scale + bb[m * 64 + 48 + n0];
      float mx = fmaxf(fmaxf(v0, v1), fmaxf(v2, v3));
#pragma unroll
      for (int off = 1; off < 16; off <<= 1) mx = fmaxf(mx, __shfl_xor(mx, off, 16));
      v0 = __expf(v0 - mx); v1 = __expf(v1 - mx);
      v2 = __expf(v2 - mx); v3 = __expf(v3 - mx);
      float sm = v0 + v1 + v2 + v3;
#pragma unroll
      for (int off = 1; off < 16; off <<= 1) sm += __shfl_xor(sm, off, 16);
      const float inv = 1.0f / sm;
      Pl[m * 64 + 0 + n0]  = (_Float16)(v0 * inv);
      Pl[m * 64 + 16 + n0] = (_Float16)(v1 * inv);
      Pl[m * 64 + 32 + n0] = (_Float16)(v2 * inv);
      Pl[m * 64 + 48 + n0] = (_Float16)(v3 * inv);
    }
  }

  // O = P V  (64x32, K=64 -> two chunks)
  v8f o[4][2];
#pragma unroll
  for (int mt = 0; mt < 4; ++mt) { o[mt][0] = vzero; o[mt][1] = vzero; }
#pragma unroll
  for (int kc = 0; kc < 2; ++kc) {
    v16h pa[4], vb[2];
#pragma unroll
    for (int mt = 0; mt < 4; ++mt) pa[mt] = load_frag(&Pl[mt * 16 * 64 + kc * 32], 64);
#pragma unroll
    for (int nt = 0; nt < 2; ++nt) vb[nt] = load_frag(&Vt[nt * 16 * 64 + kc * 32], 64);
#pragma unroll
    for (int mt = 0; mt < 4; ++mt)
#pragma unroll
      for (int nt = 0; nt < 2; ++nt)
        o[mt][nt] = __builtin_amdgcn_wmma_f32_16x16x32_f16(false, pa[mt], false, vb[nt],
                                                           (short)0, o[mt][nt], false, false);
  }

#pragma unroll
  for (int mt = 0; mt < 4; ++mt)
#pragma unroll
    for (int nt = 0; nt < 2; ++nt)
#pragma unroll
      for (int r = 0; r < 8; ++r) {
        const int t = mt * 16 + mb + r;
        const int d = nt * 16 + n0;
        outp[((size_t)win * 64 + t) * 192 + head * 32 + d] = (_Float16)o[mt][nt][r];
      }
}

// un-window + roll(+4,+4) + residual: high += gamma_main * proj_out
__global__ __launch_bounds__(256) void k_addhigh(float* __restrict__ high,
                                                 const _Float16* __restrict__ po,
                                                 const float* __restrict__ gamma) {
  const int idx = blockIdx.x * blockDim.x + threadIdx.x;
  if (idx >= 4 * 192 * 128 * 128) return;
  const int per_b = 192 * 16384;
  const int b = idx / per_b;
  const int rem = idx % per_b;
  const int c = rem >> 14;
  const int ij = rem & 16383;
  const int i = ij >> 7, j = ij & 127;
  const int iw = (i + 124) & 127, jw = (j + 124) & 127;
  const int win = (b << 8) | ((iw >> 3) << 4) | (jw >> 3);
  const int tok = ((iw & 7) << 3) | (jw & 7);
  high[idx] += gamma[0] * (float)po[((size_t)win * 64 + tok) * 192 + c];
}

// aux branch: gelu(dwconv3(ll)) -> f16 [pix][64]
__global__ __launch_bounds__(256) void k_auxdw(const float* __restrict__ ll,
                                               const float* __restrict__ dw,
                                               _Float16* __restrict__ t16) {
  const int idx = blockIdx.x * blockDim.x + threadIdx.x;
  if (idx >= 4 * 64 * 128 * 128) return;
  const int j = idx & 127, i = (idx >> 7) & 127, c = (idx >> 14) & 63, b = idx >> 20;
  const float* p = ll + (size_t)(b * 64 + c) * 16384;
  float s = 0.0f;
#pragma unroll
  for (int di = -1; di <= 1; ++di)
#pragma unroll
    for (int dj = -1; dj <= 1; ++dj) {
      const int ii = i + di, jj = j + dj;
      if (ii >= 0 && ii < 128 && jj >= 0 && jj < 128)
        s += p[ii * 128 + jj] * dw[c * 9 + (di + 1) * 3 + (dj + 1)];
    }
  t16[((size_t)((b * 128 + i) * 128 + j)) * 64 + c] = (_Float16)gelu_f(s);
}

// gap[b][c] = mean over 128x128 of ll
__global__ __launch_bounds__(256) void k_gap(const float* __restrict__ ll,
                                             float* __restrict__ gap) {
  __shared__ float red[256];
  const int bc = blockIdx.x;
  float s = 0.0f;
  for (int t = threadIdx.x; t < 16384; t += 256) s += ll[(size_t)bc * 16384 + t];
  red[threadIdx.x] = s;
  __syncthreads();
  for (int st = 128; st > 0; st >>= 1) {
    if ((int)threadIdx.x < st) red[threadIdx.x] += red[threadIdx.x + st];
    __syncthreads();
  }
  if (threadIdx.x == 0) gap[bc] = red[0] * (1.0f / 16384.0f);
}

// gate[b][0..191] = sigmoid(gelu(gap @ w1^T) @ w2^T)
__global__ __launch_bounds__(192) void k_gate(const float* __restrict__ gap,
                                              const float* __restrict__ w1,
                                              const float* __restrict__ w2,
                                              float* __restrict__ gate) {
  __shared__ float g[64];
  __shared__ float t[48];
  const int b = blockIdx.x;
  const int tid = threadIdx.x;
  if (tid < 64) g[tid] = gap[b * 64 + tid];
  __syncthreads();
  if (tid < 48) {
    float s = 0.0f;
    for (int c = 0; c < 64; ++c) s += g[c] * w1[tid * 64 + c];
    t[tid] = gelu_f(s);
  }
  __syncthreads();
  float s = 0.0f;
  for (int j = 0; j < 48; ++j) s += t[j] * w2[tid * 48 + j];
  gate[b * 192 + tid] = 1.0f / (1.0f + __expf(-s));
}

// refined_ll = ll + gamma_aux * aux (in place)
__global__ __launch_bounds__(256) void k_refinell(float* __restrict__ ll,
                                                  const float* __restrict__ auxf,
                                                  const float* __restrict__ gamma) {
  const int idx = blockIdx.x * blockDim.x + threadIdx.x;
  if (idx >= 4 * 64 * 128 * 128) return;
  const int j = idx & 127, i = (idx >> 7) & 127, c = (idx >> 14) & 63, b = idx >> 20;
  const size_t pix = (size_t)(b * 128 + i) * 128 + j;
  ll[idx] += gamma[0] * auxf[pix * 64 + c];
}

// refined_high *= 1 + gamma_cross*(gate-0.5)*2
__global__ __launch_bounds__(256) void k_gatehigh(float* __restrict__ high,
                                                  const float* __restrict__ gate,
                                                  const float* __restrict__ gamma) {
  const int idx = blockIdx.x * blockDim.x + threadIdx.x;
  if (idx >= 4 * 192 * 128 * 128) return;
  const int per_b = 192 * 16384;
  const int b = idx / per_b;
  const int c = (idx % per_b) >> 14;
  high[idx] *= 1.0f + gamma[0] * (gate[b * 192 + c] - 0.5f) * 2.0f;
}

// IWT and subtract x -> diff f16 [pix][64]
__global__ __launch_bounds__(256) void k_iwt_diff(const float* __restrict__ ll,
                                                  const float* __restrict__ high,
                                                  const float* __restrict__ x,
                                                  _Float16* __restrict__ diff) {
  const int idx = blockIdx.x * blockDim.x + threadIdx.x;
  if (idx >= 4 * 64 * 128 * 128) return;
  const int j = idx & 127, i = (idx >> 7) & 127, c = (idx >> 14) & 63, b = idx >> 20;
  const float lv = ll[idx];
  const size_t hb = (((size_t)b * 192 + c) * 128 + i) * 128 + j;
  const float hl = high[hb];
  const float lh = high[hb + 64 * 128 * 128];
  const float hh = high[hb + 128 * 128 * 128];
  const float x1 = (lv - hl - lh + hh) * 0.5f;
  const float x2 = (lv - hl + lh - hh) * 0.5f;
  const float x3 = (lv + hl - lh - hh) * 0.5f;
  const float x4 = (lv + hl + lh + hh) * 0.5f;
  const size_t xb = ((size_t)(b * 64 + c)) * 65536;
  const int i2 = 2 * i, j2 = 2 * j;
  const size_t p00 = (size_t)b * 65536 + i2 * 256 + j2;
  diff[p00 * 64 + c]         = (_Float16)(x1 - x[xb + i2 * 256 + j2]);
  diff[(p00 + 1) * 64 + c]   = (_Float16)(x3 - x[xb + i2 * 256 + j2 + 1]);
  diff[(p00 + 256) * 64 + c] = (_Float16)(x2 - x[xb + (i2 + 1) * 256 + j2]);
  diff[(p00 + 257) * 64 + c] = (_Float16)(x4 - x[xb + (i2 + 1) * 256 + j2 + 1]);
}

// y = x + lhmsa; LayerNorm(y) over 64 channels -> ln16; keep y f32 [pix][64]
__global__ __launch_bounds__(64) void k_resid_ln(const float* __restrict__ x,
                                                 const float* __restrict__ lhm,
                                                 const float* __restrict__ w,
                                                 const float* __restrict__ bse,
                                                 float* __restrict__ yb,
                                                 _Float16* __restrict__ ln16) {
  __shared__ float red[64];
  __shared__ float smu, srs;
  const int pix = blockIdx.x;
  const int c = threadIdx.x;
  const int b = pix >> 16;
  const int ij = pix & 65535;
  const float xv = x[((size_t)(b * 64 + c)) * 65536 + ij];
  const float y = xv + lhm[(size_t)pix * 64 + c];
  yb[(size_t)pix * 64 + c] = y;
  red[c] = y;
  __syncthreads();
  for (int st = 32; st > 0; st >>= 1) { if (c < st) red[c] += red[c + st]; __syncthreads(); }
  if (c == 0) smu = red[0] * (1.0f / 64.0f);
  __syncthreads();
  const float d = y - smu;
  red[c] = d * d;
  __syncthreads();
  for (int st = 32; st > 0; st >>= 1) { if (c < st) red[c] += red[c + st]; __syncthreads(); }
  if (c == 0) srs = rsqrtf(red[0] * (1.0f / 64.0f) + 1e-6f);
  __syncthreads();
  ln16[(size_t)pix * 64 + c] = (_Float16)(d * srs * w[c] + bse[c]);
}

// depthwise 3x3 conv on h16 [b][i][j][256] -> hdw16
__global__ __launch_bounds__(256) void k_ffndw(const _Float16* __restrict__ h16,
                                               const float* __restrict__ dw,
                                               _Float16* __restrict__ hdw) {
  const size_t idx = (size_t)blockIdx.x * blockDim.x + threadIdx.x;
  if (idx >= (size_t)262144 * 256) return;
  const int ch = (int)(idx & 255);
  const int pix = (int)(idx >> 8);
  const int b = pix >> 16;
  const int ij = pix & 65535;
  const int i = ij >> 8, j = ij & 255;
  float s = 0.0f;
  const float* wk = dw + ch * 9;
#pragma unroll
  for (int di = -1; di <= 1; ++di)
#pragma unroll
    for (int dj = -1; dj <= 1; ++dj) {
      const int ii = i + di, jj = j + dj;
      if (ii >= 0 && ii < 256 && jj >= 0 && jj < 256)
        s += (float)h16[((size_t)(b * 65536 + ii * 256 + jj)) * 256 + ch] *
             wk[(di + 1) * 3 + (dj + 1)];
    }
  hdw[idx] = (_Float16)s;
}

// spectral weights -> spatial 8x8 circular-conv kernels: ks[ch][r][s]=irfft2(W)
__global__ void k_spec(const float* __restrict__ wr, const float* __restrict__ wi,
                       float* __restrict__ ks) {
  const int t = blockIdx.x * blockDim.x + threadIdx.x;
  if (t >= 128 * 64) return;
  const int ch = t >> 6;
  const int r = (t >> 3) & 7, s = t & 7;
  float acc = 0.0f;
  for (int u = 0; u < 8; ++u)
    for (int v = 0; v < 5; ++v) {
      const float cv = (v == 0 || v == 4) ? 1.0f : 2.0f;
      const float ang = 0.78539816339744831f * (float)(u * r + v * s);  // 2*pi/8
      const float cr = wr[ch * 40 + u * 5 + v];
      const float ci = wi[ch * 40 + u * 5 + v];
      acc += cv * (cr * __cosf(ang) - ci * __sinf(ang));
    }
  ks[t] = acc * (1.0f / 64.0f);
}

// per-patch 8x8 circular convolution of x1 + gating: g = gelu(conv(x1)) * x2
__global__ __launch_bounds__(256) void k_patchconv(const _Float16* __restrict__ hdw,
                                                   const float* __restrict__ ks,
                                                   _Float16* __restrict__ g16) {
  const size_t idx = (size_t)blockIdx.x * blockDim.x + threadIdx.x;
  if (idx >= (size_t)262144 * 128) return;
  const int ch = (int)(idx & 127);
  const int pix = (int)(idx >> 7);
  const int b = pix >> 16;
  const int ij = pix & 65535;
  const int i = ij >> 8, j = ij & 255;
  const int bi = i & ~7, bj = j & ~7;
  const int r = i & 7, s = j & 7;
  float acc = 0.0f;
  const float* kk = ks + ch * 64;
#pragma unroll
  for (int r2 = 0; r2 < 8; ++r2) {
    const size_t rowb = ((size_t)(b * 65536 + (bi + r2) * 256 + bj)) * 256 + ch;
    const int kr = ((r - r2) & 7) * 8;
#pragma unroll
    for (int s3 = 0; s3 < 8; ++s3)
      acc += (float)hdw[rowb + (size_t)s3 * 256] * kk[kr + ((s - s3) & 7)];
  }
  const float x2v = (float)hdw[(size_t)pix * 256 + 128 + ch];
  g16[(size_t)pix * 128 + ch] = (_Float16)(gelu_f(acc) * x2v);
}

// out (b,c,i,j) = y + ffn_out
__global__ __launch_bounds__(256) void k_final(const float* __restrict__ yb,
                                               const float* __restrict__ fo,
                                               float* __restrict__ out) {
  const int idx = blockIdx.x * blockDim.x + threadIdx.x;
  if (idx >= 4 * 64 * 65536) return;
  const int ij = idx & 65535;
  const int c = (idx >> 16) & 63;
  const int b = idx >> 22;
  const size_t pix = (size_t)b * 65536 + ij;
  out[idx] = yb[pix * 64 + c] + fo[pix * 64 + c];
}

// ---------------------------------------------------------------------------
// Host launcher
// ---------------------------------------------------------------------------
extern "C" void kernel_launch(void* const* d_in, const int* in_sizes, int n_in,
                              void* d_out, int out_size, void* d_ws, size_t ws_size,
                              hipStream_t stream) {
  (void)in_sizes; (void)n_in; (void)out_size; (void)ws_size;
  const float* x          = (const float*)d_in[0];
  const float* ln_w       = (const float*)d_in[1];
  const float* ln_b       = (const float*)d_in[2];
  const float* lnh_w      = (const float*)d_in[3];
  const float* lnh_b      = (const float*)d_in[4];
  const float* w_qkv      = (const float*)d_in[5];
  const float* b_qkv      = (const float*)d_in[6];
  const float* w_proj     = (const float*)d_in[7];
  const float* b_proj     = (const float*)d_in[8];
  const float* rel_table  = (const float*)d_in[9];
  const float* gamma_main = (const float*)d_in[10];
  const float* gamma_aux  = (const float*)d_in[11];
  const float* gamma_crs  = (const float*)d_in[12];
  const float* aux_dw     = (const float*)d_in[13];
  const float* aux_pw     = (const float*)d_in[14];
  const float* cg_w1      = (const float*)d_in[15];
  const float* cg_w2      = (const float*)d_in[16];
  const float* attn_out_w = (const float*)d_in[17];
  const float* ffn_in_w   = (const float*)d_in[18];
  const float* ffn_dw     = (const float*)d_in[19];
  const float* ffn_out_w  = (const float*)d_in[20];
  const float* w_base_r   = (const float*)d_in[21];
  const float* w_base_i   = (const float*)d_in[22];
  const int*   rel_idx    = (const int*)d_in[23];
  float* out = (float*)d_out;

  constexpr size_t MiB = 1ull << 20;
  char* w = (char*)d_ws;
  // Region G: wavelet bands + small tables/packed weights
  float*    ll     = (float*)(w + 0);                 // 16 MiB
  float*    high   = (float*)(w + 16 * MiB);          // 48 MiB
  float*    bias66 = (float*)(w + 64 * MiB);          // 96 KiB
  float*    kspat  = (float*)(w + 64 * MiB + 128 * 1024);
  float*    gapb   = (float*)(w + 64 * MiB + 192 * 1024);
  float*    gateb  = (float*)(w + 64 * MiB + 196 * 1024);
  _Float16* pqkv   = (_Float16*)(w + 64 * MiB + 256 * 1024);
  _Float16* pproj  = (_Float16*)(w + 64 * MiB + 512 * 1024);
  _Float16* paux   = (_Float16*)(w + 64 * MiB + 640 * 1024);
  _Float16* pao    = (_Float16*)(w + 64 * MiB + 704 * 1024);
  _Float16* pfi    = (_Float16*)(w + 64 * MiB + 768 * 1024);
  _Float16* pfo    = (_Float16*)(w + 64 * MiB + 832 * 1024);
  // Region A (128 MiB): hn16+qkv16 early, reused by h16 later
  _Float16* hn16  = (_Float16*)(w + 65 * MiB);
  _Float16* qkv16 = (_Float16*)(w + 89 * MiB);        // 65+24
  _Float16* h16   = (_Float16*)(w + 65 * MiB);        // reuse
  // Region B (128 MiB): attention/aux intermediates, reused by hdw16
  _Float16* att16 = (_Float16*)(w + 193 * MiB);
  _Float16* po16  = (_Float16*)(w + 217 * MiB);
  _Float16* t16   = (_Float16*)(w + 241 * MiB);
  float*    auxf  = (float*)(w + 249 * MiB);
  _Float16* hdw16 = (_Float16*)(w + 193 * MiB);       // reuse
  // Region C (64 MiB): diff16+ln16 early, reused by g16
  _Float16* diff16 = (_Float16*)(w + 321 * MiB);
  _Float16* ln16   = (_Float16*)(w + 353 * MiB);
  _Float16* g16    = (_Float16*)(w + 321 * MiB);      // reuse
  // Region D (64 MiB): lhm then ffnout
  float* lhm    = (float*)(w + 385 * MiB);
  float* ffnout = (float*)(w + 385 * MiB);            // reuse
  // Region E (64 MiB): y residual
  float* yb = (float*)(w + 449 * MiB);

  auto pack = [&](const float* W, _Float16* Bp, int K, int N) {
    const int total = (K / 32) * (N / 16) * 512;
    k_pack_b<<<(total + 255) / 256, 256, 0, stream>>>(W, Bp, K, N);
  };
  auto gemm = [&](const _Float16* A, const _Float16* Bp, const float* bias,
                  float* Cf, _Float16* Ch, int M, int N, int K) {
    const int strips = (M / 16) * (N / 64);
    k_gemm<<<strips / 8, 256, 0, stream>>>(A, Bp, bias, Cf, Ch, M, N, K);
  };

  // --- stage 0: prep ---
  k_dwt<<<(4 * 64 * 128 * 128) / 256, 256, 0, stream>>>(x, ll, high);
  k_relbias<<<(6 * 4096 + 255) / 256, 256, 0, stream>>>(rel_table, rel_idx, bias66);
  k_spec<<<(128 * 64 + 255) / 256, 256, 0, stream>>>(w_base_r, w_base_i, kspat);
  pack(w_qkv,      pqkv,  192, 576);
  pack(w_proj,     pproj, 192, 192);
  pack(aux_pw,     paux,  64,  64);
  pack(attn_out_w, pao,   64,  64);
  pack(ffn_in_w,   pfi,   64,  256);
  pack(ffn_out_w,  pfo,   128, 64);

  // --- LHMSA high-band path ---
  k_ln_hn<<<65536, 256, 0, stream>>>(high, lnh_w, lnh_b, hn16);
  gemm(hn16, pqkv, b_qkv, nullptr, qkv16, 65536, 576, 192);
  k_attn<<<dim3(1024, 6), 32, 0, stream>>>(qkv16, bias66, att16);
  gemm(att16, pproj, b_proj, nullptr, po16, 65536, 192, 192);
  k_addhigh<<<(4 * 192 * 128 * 128) / 256, 256, 0, stream>>>(high, po16, gamma_main);

  // --- aux / gate path ---
  k_auxdw<<<(4 * 64 * 128 * 128) / 256, 256, 0, stream>>>(ll, aux_dw, t16);
  gemm(t16, paux, nullptr, auxf, nullptr, 65536, 64, 64);
  k_gap<<<256, 256, 0, stream>>>(ll, gapb);
  k_gate<<<4, 192, 0, stream>>>(gapb, cg_w1, cg_w2, gateb);
  k_refinell<<<(4 * 64 * 128 * 128) / 256, 256, 0, stream>>>(ll, auxf, gamma_aux);
  k_gatehigh<<<(4 * 192 * 128 * 128) / 256, 256, 0, stream>>>(high, gateb, gamma_crs);

  // --- IWT, attn_out conv, residual + LN ---
  k_iwt_diff<<<(4 * 64 * 128 * 128) / 256, 256, 0, stream>>>(ll, high, x, diff16);
  gemm(diff16, pao, nullptr, lhm, nullptr, 262144, 64, 64);
  k_resid_ln<<<262144, 64, 0, stream>>>(x, lhm, ln_w, ln_b, yb, ln16);

  // --- PSFFN ---
  gemm(ln16, pfi, nullptr, nullptr, h16, 262144, 256, 64);
  k_ffndw<<<262144, 256, 0, stream>>>(h16, ffn_dw, hdw16);
  k_patchconv<<<131072, 256, 0, stream>>>(hdw16, kspat, g16);
  gemm(g16, pfo, nullptr, ffnout, nullptr, 262144, 64, 128);
  k_final<<<(4 * 64 * 65536) / 256, 256, 0, stream>>>(yb, ffnout, out);
}
